// CrossAlignMatrix_72395968741788
// MI455X (gfx1250) — matrix operand order repaired and checked by reading, not compiled, not run on hardware
//
#include <hip/hip_runtime.h>

typedef __attribute__((ext_vector_type(16))) __bf16 v16bf;
typedef __attribute__((ext_vector_type(8)))  __bf16 v8bf;
typedef __attribute__((ext_vector_type(8)))  float  v8f;

constexpr int B_  = 16;
constexpr int LC_ = 1024;
constexpr int LQ_ = 1024;
constexpr int D_  = 1024;

__device__ __forceinline__ unsigned short f2bf(float x) {
  unsigned u = __builtin_bit_cast(unsigned, x);
  unsigned r = (u + 0x7FFFu + ((u >> 16) & 1u)) >> 16;  // round-to-nearest-even
  return (unsigned short)r;
}

__device__ __forceinline__ v8f zero8() {
  v8f z;
#pragma unroll
  for (int i = 0; i < 8; ++i) z[i] = 0.f;
  return z;
}

// A-fragment for V_WMMA_*_16X16X32_BF16: lane&15 = M row; the lane holds
// K = [k0+16*half, +16) in a[0..15] (same contiguous order as the B loads)
// -> two contiguous 16-byte chunks.
__device__ __forceinline__ v16bf loadA(const __bf16* p, int half) {
  const v8bf a0 = *(const v8bf*)(p + 16 * half);
  const v8bf a1 = *(const v8bf*)(p + 16 * half + 8);
  return __builtin_shufflevector(a0, a1, 0, 1, 2, 3, 4, 5, 6, 7,
                                 8, 9, 10, 11, 12, 13, 14, 15);
}

// ---------------------------------------------------------------------------
// Kernel P: bf16 conversions + row-dot reductions.
//   rows [0, B*LC)          : cb = bf16(c * w_cq), sc = c.w_c + b_c + b_cq
//   rows [B*LC, B*LC+B*LQ)  : qb = bf16(q),        sq = q.w_q + b_q
//   rows [.., .. + D)       : wb = bf16(W_out)
// ---------------------------------------------------------------------------
__global__ void prep_kernel(const float* __restrict__ c, const float* __restrict__ q,
                            const float* __restrict__ w_c, const float* __restrict__ b_c,
                            const float* __restrict__ w_q, const float* __restrict__ b_q,
                            const float* __restrict__ w_cq, const float* __restrict__ b_cq,
                            const float* __restrict__ W_out,
                            unsigned short* __restrict__ cb, unsigned short* __restrict__ qb,
                            unsigned short* __restrict__ wb,
                            float* __restrict__ sc, float* __restrict__ sq) {
  __shared__ float red[256];
  const int row = blockIdx.x;
  const int tid = threadIdx.x;

  if (row < B_ * LC_) {
    const float* src = c + (size_t)row * D_;
    float part = 0.f;
#pragma unroll
    for (int e = 0; e < 4; ++e) {
      const int d = tid * 4 + e;
      const float v = src[d];
      part += v * w_c[d];
      cb[(size_t)row * D_ + d] = f2bf(v * w_cq[d]);
    }
    red[tid] = part;
    __syncthreads();
    for (int s = 128; s > 0; s >>= 1) {
      if (tid < s) red[tid] += red[tid + s];
      __syncthreads();
    }
    if (tid == 0) sc[row] = red[0] + b_c[0] + b_cq[0];
  } else if (row < B_ * LC_ + B_ * LQ_) {
    const int r2 = row - B_ * LC_;
    const float* src = q + (size_t)r2 * D_;
    float part = 0.f;
#pragma unroll
    for (int e = 0; e < 4; ++e) {
      const int d = tid * 4 + e;
      const float v = src[d];
      part += v * w_q[d];
      qb[(size_t)r2 * D_ + d] = f2bf(v);
    }
    red[tid] = part;
    __syncthreads();
    for (int s = 128; s > 0; s >>= 1) {
      if (tid < s) red[tid] += red[tid + s];
      __syncthreads();
    }
    if (tid == 0) sq[r2] = red[0] + b_q[0];
  } else {
    const int e = row - (B_ * LC_ + B_ * LQ_);
    const float* src = W_out + (size_t)e * D_;
#pragma unroll
    for (int k = 0; k < 4; ++k) {
      const int d = tid * 4 + k;
      wb[(size_t)e * D_ + d] = f2bf(src[d]);
    }
  }
}

// ---------------------------------------------------------------------------
// Kernel T: qT[b,d,j] = bf16(q[b,j,d]), LDS-tiled 64x64 transpose (coalesced
// both directions). Makes the a@q GEMM's B-fragment a contiguous 32-byte run.
// ---------------------------------------------------------------------------
__global__ void transpose_kernel(const float* __restrict__ q, unsigned short* __restrict__ qT) {
  __shared__ unsigned short tile[64][72];  // padded to dodge bank conflicts
  const int b = blockIdx.z;
  const int j0 = blockIdx.x * 64;
  const int d0 = blockIdx.y * 64;
  const int tx = threadIdx.x & 63;
  const int ty = threadIdx.x >> 6;  // 0..3
  const float* qsrc = q + (size_t)b * LQ_ * D_;
#pragma unroll
  for (int rr = 0; rr < 16; ++rr) {
    const int j = ty * 16 + rr;
    tile[j][tx] = f2bf(qsrc[(size_t)(j0 + j) * D_ + d0 + tx]);
  }
  __syncthreads();
  unsigned short* dst = qT + (size_t)b * D_ * LQ_;
#pragma unroll
  for (int rr = 0; rr < 16; ++rr) {
    const int d = ty * 16 + rr;
    dst[(size_t)(d0 + d) * LQ_ + j0 + tx] = tile[tx][d];
  }
}

// ---------------------------------------------------------------------------
// Kernel S: scores + fused softmax.
// Block = 256 threads (8 waves). Tile: 16 rows of c x full LQ.
// Wave w owns columns [w*128, w*128+128) = 8 WMMA N-tiles; A-frag is reused
// across all 8 WMMAs per K-step.
// ---------------------------------------------------------------------------
__global__ void score_softmax_kernel(const __bf16* __restrict__ cb, const __bf16* __restrict__ qb,
                                     const float* __restrict__ sc, const float* __restrict__ sq,
                                     const int* __restrict__ q_mask,
                                     unsigned short* __restrict__ a_out) {
  const int b = blockIdx.y;
  const int i0 = blockIdx.x * 16;
  const int tid = threadIdx.x;
  const int wave = tid >> 5;
  const int lane = tid & 31;
  const int half = lane >> 4;
  const int nl = lane & 15;
  const int j0 = wave * 128;

  const __bf16* Arow = cb + (size_t)(b * LC_ + i0 + nl) * D_;
  const __bf16* Bbat = qb + (size_t)b * LQ_ * D_;

  v8f acc[8];
#pragma unroll
  for (int t = 0; t < 8; ++t) acc[t] = zero8();

  for (int kt = 0; kt < D_ / 32; ++kt) {
    const int k0 = kt * 32;
    const v16bf af = loadA(Arow + k0, half);
#pragma unroll
    for (int t = 0; t < 8; ++t) {
      const __bf16* Brow = Bbat + (size_t)(j0 + t * 16 + nl) * D_;
      const v16bf bfv = *(const v16bf*)(Brow + k0 + 16 * half);
      acc[t] = __builtin_amdgcn_wmma_f32_16x16x32_bf16(false, af, false, bfv,
                                                       (short)0, acc[t], false, false);
    }
  }

  // s = sc[i] + sq[j] + cq ; clip ; mask
  float scv[8];
#pragma unroll
  for (int r = 0; r < 8; ++r) scv[r] = sc[b * LC_ + i0 + r + 8 * half];

#pragma unroll
  for (int t = 0; t < 8; ++t) {
    const int j = j0 + t * 16 + nl;
    const float sqv = sq[b * LQ_ + j];
    const int mk = q_mask[b * LQ_ + j];
#pragma unroll
    for (int r = 0; r < 8; ++r) {
      float s = acc[t][r] + scv[r] + sqv;
      s = fminf(15.f, fmaxf(-15.f, s));
      if (mk == 0) s = -1e30f;
      acc[t][r] = s;
    }
  }

  __shared__ float redMax[8][16];
  __shared__ float redSum[8][16];
  __shared__ float gmax[16];
  __shared__ float gsum[16];

  // row max: local N-tiles -> 16 lanes (xor 1..8 stays inside the half) -> waves.
  float mrow[8];
#pragma unroll
  for (int r = 0; r < 8; ++r) mrow[r] = acc[0][r];
#pragma unroll
  for (int t = 1; t < 8; ++t)
#pragma unroll
    for (int r = 0; r < 8; ++r) mrow[r] = fmaxf(mrow[r], acc[t][r]);
#pragma unroll
  for (int m = 1; m <= 8; m <<= 1)
#pragma unroll
    for (int r = 0; r < 8; ++r) mrow[r] = fmaxf(mrow[r], __shfl_xor(mrow[r], m, 32));
  if (nl == 0) {
#pragma unroll
    for (int r = 0; r < 8; ++r) redMax[wave][r + 8 * half] = mrow[r];
  }
  __syncthreads();
  if (tid < 16) {
    float g = -3.0e38f;
#pragma unroll
    for (int w = 0; w < 8; ++w) g = fmaxf(g, redMax[w][tid]);
    gmax[tid] = g;
  }
  __syncthreads();
#pragma unroll
  for (int r = 0; r < 8; ++r) mrow[r] = gmax[r + 8 * half];

  // exp + row sum
  float srow[8];
#pragma unroll
  for (int r = 0; r < 8; ++r) srow[r] = 0.f;
#pragma unroll
  for (int t = 0; t < 8; ++t)
#pragma unroll
    for (int r = 0; r < 8; ++r) {
      const float e = __expf(acc[t][r] - mrow[r]);
      acc[t][r] = e;
      srow[r] += e;
    }
#pragma unroll
  for (int m = 1; m <= 8; m <<= 1)
#pragma unroll
    for (int r = 0; r < 8; ++r) srow[r] += __shfl_xor(srow[r], m, 32);
  if (nl == 0) {
#pragma unroll
    for (int r = 0; r < 8; ++r) redSum[wave][r + 8 * half] = srow[r];
  }
  __syncthreads();
  if (tid < 16) {
    float g = 0.f;
#pragma unroll
    for (int w = 0; w < 8; ++w) g += redSum[w][tid];
    gsum[tid] = g;
  }
  __syncthreads();

  float inv[8];
#pragma unroll
  for (int r = 0; r < 8; ++r) inv[r] = 1.0f / gsum[r + 8 * half];

#pragma unroll
  for (int t = 0; t < 8; ++t)
#pragma unroll
    for (int r = 0; r < 8; ++r)
      a_out[(size_t)(b * LC_ + i0 + r + 8 * half) * LQ_ + j0 + t * 16 + nl] =
          f2bf(acc[t][r] * inv[r]);
}

// ---------------------------------------------------------------------------
// Register-blocked bf16 GEMM: out[b, i, n] = sum_k A[b, i, k] * Bt[b, n, k]
// M = N = K = 1024. Wave computes a 32x64 block: 2 M-tiles x 4 N-tiles
// -> 8 WMMAs per K-step fed by 12 b128 loads (1.5 loads/WMMA instead of 4).
// Block = 8 waves covering 32 rows x 512 cols; grid = (2, 32, B).
// f32_out=1: write f32 + bias[n]; else write bf16.
// ---------------------------------------------------------------------------
__global__ void gemm_bf16_kernel(const __bf16* __restrict__ A, size_t a_bstride,
                                 const __bf16* __restrict__ Bt, size_t b_bstride,
                                 const float* __restrict__ bias,
                                 unsigned short* __restrict__ outb,
                                 float* __restrict__ outf, int f32_out,
                                 size_t o_bstride) {
  const int b = blockIdx.z;
  const int i0 = blockIdx.y * 32;
  const int tid = threadIdx.x;
  const int wave = tid >> 5;
  const int lane = tid & 31;
  const int half = lane >> 4;
  const int nl = lane & 15;
  const int n0 = blockIdx.x * 512 + wave * 64;

  const __bf16* Arow0 = A + b * a_bstride + (size_t)(i0 + nl) * 1024;
  const __bf16* Arow1 = Arow0 + (size_t)16 * 1024;
  const __bf16* Bb = Bt + b * b_bstride;

  v8f acc[2][4];
#pragma unroll
  for (int mt = 0; mt < 2; ++mt)
#pragma unroll
    for (int nt = 0; nt < 4; ++nt) acc[mt][nt] = zero8();

  for (int kt = 0; kt < 1024 / 32; ++kt) {
    const int k0 = kt * 32;
    const v16bf af0 = loadA(Arow0 + k0, half);
    const v16bf af1 = loadA(Arow1 + k0, half);
#pragma unroll
    for (int nt = 0; nt < 4; ++nt) {
      const __bf16* Brow = Bb + (size_t)(n0 + nt * 16 + nl) * 1024;
      const v16bf bfv = *(const v16bf*)(Brow + k0 + 16 * half);
      acc[0][nt] = __builtin_amdgcn_wmma_f32_16x16x32_bf16(false, af0, false, bfv,
                                                           (short)0, acc[0][nt], false, false);
      acc[1][nt] = __builtin_amdgcn_wmma_f32_16x16x32_bf16(false, af1, false, bfv,
                                                           (short)0, acc[1][nt], false, false);
    }
  }

#pragma unroll
  for (int nt = 0; nt < 4; ++nt) {
    const int col = n0 + nt * 16 + nl;
    const float bo = f32_out ? bias[col] : 0.f;
#pragma unroll
    for (int mt = 0; mt < 2; ++mt)
#pragma unroll
      for (int r = 0; r < 8; ++r) {
        const size_t idx = b * o_bstride + (size_t)(i0 + mt * 16 + r + 8 * half) * 1024 + col;
        if (f32_out) outf[idx] = acc[mt][nt][r] + bo;
        else         outb[idx] = f2bf(acc[mt][nt][r]);
      }
  }
}

// ---------------------------------------------------------------------------
extern "C" void kernel_launch(void* const* d_in, const int* in_sizes, int n_in,
                              void* d_out, int out_size, void* d_ws, size_t ws_size,
                              hipStream_t stream) {
  (void)in_sizes; (void)n_in; (void)out_size; (void)ws_size;
  const float* c     = (const float*)d_in[0];
  const float* q     = (const float*)d_in[1];
  const int*   qmask = (const int*)d_in[2];
  const float* w_c   = (const float*)d_in[3];
  const float* b_c   = (const float*)d_in[4];
  const float* w_q   = (const float*)d_in[5];
  const float* b_q   = (const float*)d_in[6];
  const float* w_cq  = (const float*)d_in[7];
  const float* b_cq  = (const float*)d_in[8];
  const float* W_out = (const float*)d_in[9];
  const float* b_out = (const float*)d_in[10];
  float* out = (float*)d_out;

  char* ws = (char*)d_ws;
  size_t off = 0;
  auto take = [&](size_t bytes) -> void* {
    void* p = ws + off;
    off += (bytes + 255) & ~(size_t)255;
    return p;
  };
  unsigned short* cb  = (unsigned short*)take((size_t)B_ * LC_ * D_ * 2);
  unsigned short* qb  = (unsigned short*)take((size_t)B_ * LQ_ * D_ * 2);
  unsigned short* qT  = (unsigned short*)take((size_t)B_ * D_ * LQ_ * 2);
  unsigned short* wb  = (unsigned short*)take((size_t)D_ * D_ * 2);
  unsigned short* aat = (unsigned short*)take((size_t)B_ * LC_ * LQ_ * 2);
  unsigned short* c2q = (unsigned short*)take((size_t)B_ * LC_ * D_ * 2);
  float* sc = (float*)take((size_t)B_ * LC_ * 4);
  float* sq = (float*)take((size_t)B_ * LQ_ * 4);

  const int prep_blocks = B_ * LC_ + B_ * LQ_ + D_;
  prep_kernel<<<prep_blocks, 256, 0, stream>>>(c, q, w_c, b_c, w_q, b_q, w_cq, b_cq,
                                               W_out, cb, qb, wb, sc, sq);

  transpose_kernel<<<dim3(LQ_ / 64, D_ / 64, B_), 256, 0, stream>>>(q, qT);

  score_softmax_kernel<<<dim3(LC_ / 16, B_), 256, 0, stream>>>(
      (const __bf16*)cb, (const __bf16*)qb, sc, sq, qmask, aat);

  // c2q = a @ q    (A = a [LC x LQ], Bt = qT [D x LQ], bf16 out)
  gemm_bf16_kernel<<<dim3(2, 32, B_), 256, 0, stream>>>(
      (const __bf16*)aat, (size_t)LC_ * LQ_,
      (const __bf16*)qT, (size_t)D_ * LQ_,
      b_out, c2q, nullptr, 0, (size_t)LC_ * D_);

  // out = c2q @ W_out^T + b_out  (Bt = wb [D x D], batch stride 0, f32 out)
  gemm_bf16_kernel<<<dim3(2, 32, B_), 256, 0, stream>>>(
      (const __bf16*)c2q, (size_t)LC_ * D_,
      (const __bf16*)wb, (size_t)0,
      b_out, nullptr, out, 1, (size_t)LC_ * D_);
}
